// GATAE_73521250173074
// MI455X (gfx1250) — matrix operand order repaired and checked by reading, not compiled, not run on hardware
//
#include <hip/hip_runtime.h>
#include <hip/hip_fp16.h>

// ---------------------------------------------------------------------------
// GATv2 autoencoder forward for MI455X (gfx1250, wave32).
// Dominant cost: A_pred = sigmoid(z @ z^T) -> 256MB f32 store stream
// (store-bound at 23.3 TB/s, ~11us). K=32 matches V_WMMA_F32_16X16X32_F16
// exactly, so each 16x16 tile is one WMMA; z (512KB) is L2-resident.
// Epilogue uses v_exp + v_rcp (approx) and nontemporal b32 stores so VALU
// and L2 stay off the critical path.
// ---------------------------------------------------------------------------

#define N_NODES 8192
#define E_EDGES 262144
#define ETOT    (E_EDGES + N_NODES)   // edges + self loops = 270336
#define IN_F    32
#define HID     16
#define HEADS   8
#define F1      (HEADS * HID)          // 128
#define OUT_F   32
#define SLOPE   0.2f

typedef _Float16 v16h __attribute__((ext_vector_type(16)));
typedef _Float16 v8h  __attribute__((ext_vector_type(8)));
typedef float    v8f  __attribute__((ext_vector_type(8)));

// ---- ordered-uint encoding of float for atomicMax ----
__device__ __forceinline__ unsigned fkey(float f) {
    unsigned u = __float_as_uint(f);
    return (u & 0x80000000u) ? ~u : (u | 0x80000000u);
}
__device__ __forceinline__ float fdecode(unsigned k) {
    unsigned u = (k & 0x80000000u) ? (k & 0x7FFFFFFFu) : ~k;
    return __uint_as_float(u);
}
__device__ __forceinline__ float lrelu(float v) { return v > 0.0f ? v : SLOPE * v; }

// fast sigmoid: v_exp_f32 + v_add + v_rcp_f32 (no IEEE div expansion)
__device__ __forceinline__ float fast_sigmoid(float x) {
    return __builtin_amdgcn_rcpf(1.0f + __expf(-x));
}

__device__ __forceinline__ void edge_pair(const int* __restrict__ ei, int e, int& s, int& d) {
    if (e < E_EDGES) { s = ei[e]; d = ei[E_EDGES + e]; }
    else             { s = e - E_EDGES; d = s; }
}

// ---------------- conv layer 1: linear transforms ----------------
__global__ void k_linear1(const float* __restrict__ x,
                          const float* __restrict__ Wl, const float* __restrict__ bl,
                          const float* __restrict__ Wr, const float* __restrict__ br,
                          float* __restrict__ xl, float* __restrict__ xr) {
    int t = blockIdx.x * blockDim.x + threadIdx.x;     // N * F1 threads
    int n = t >> 7, c = t & 127;
    const float* xp = x + n * IN_F;
    float al = bl[c], ar = br[c];
#pragma unroll
    for (int k = 0; k < IN_F; ++k) {
        float xv = xp[k];
        al = fmaf(xv, Wl[k * F1 + c], al);
        ar = fmaf(xv, Wr[k * F1 + c], ar);
    }
    xl[t] = al; xr[t] = ar;
}

// ---------------- conv1 edge pass A: scores + segment max ----------------
__global__ void k_score1(const int* __restrict__ ei,
                         const float* __restrict__ xl, const float* __restrict__ xr,
                         const float* __restrict__ att,
                         float* __restrict__ e1, unsigned* __restrict__ m1) {
    int t = blockIdx.x * blockDim.x + threadIdx.x;     // ETOT * HEADS threads
    int e = t >> 3, h = t & 7;
    int s, d; edge_pair(ei, e, s, d);
    const float* pl = xl + s * F1 + h * HID;
    const float* pr = xr + d * F1 + h * HID;
    const float* pa = att + h * HID;
    float acc = 0.0f;
#pragma unroll
    for (int c = 0; c < HID; ++c)
        acc = fmaf(pa[c], lrelu(pl[c] + pr[c]), acc);
    e1[t] = acc;
    atomicMax(&m1[d * HEADS + h], fkey(acc));
}

// ---------------- conv1 edge pass B: exp + segment sum ----------------
__global__ void k_exp1(const int* __restrict__ ei,
                       float* __restrict__ e1, const unsigned* __restrict__ m1,
                       float* __restrict__ s1) {
    int t = blockIdx.x * blockDim.x + threadIdx.x;
    int e = t >> 3, h = t & 7;
    int s, d; edge_pair(ei, e, s, d);
    (void)s;
    float p = __expf(e1[t] - fdecode(m1[d * HEADS + h]));
    e1[t] = p;
    atomicAdd(&s1[d * HEADS + h], p);
}

// ---------------- conv1 edge pass C: weighted aggregation ----------------
__global__ void k_agg1(const int* __restrict__ ei,
                       const float* __restrict__ e1, const float* __restrict__ s1,
                       const float* __restrict__ xl, float* __restrict__ h1) {
    int t = blockIdx.x * blockDim.x + threadIdx.x;
    int e = t >> 3, h = t & 7;
    int s, d; edge_pair(ei, e, s, d);
    float alpha = e1[t] / (s1[d * HEADS + h] + 1e-16f);
    const float* pl = xl + s * F1 + h * HID;
    float* po = h1 + d * F1 + h * HID;
#pragma unroll
    for (int c = 0; c < HID; ++c)
        atomicAdd(&po[c], alpha * pl[c]);
}

// ---------------- conv layer 2: linear transforms (bias1 folded in) --------
__global__ void k_linear2(const float* __restrict__ h1, const float* __restrict__ bias1,
                          const float* __restrict__ Wl, const float* __restrict__ bl,
                          const float* __restrict__ Wr, const float* __restrict__ br,
                          float* __restrict__ xl, float* __restrict__ xr) {
    int t = blockIdx.x * blockDim.x + threadIdx.x;     // N * OUT_F threads
    int n = t >> 5, c = t & 31;
    const float* hp = h1 + n * F1;
    float al = bl[c], ar = br[c];
#pragma unroll 8
    for (int k = 0; k < F1; ++k) {
        float hv = hp[k] + bias1[k];
        al = fmaf(hv, Wl[k * OUT_F + c], al);
        ar = fmaf(hv, Wr[k * OUT_F + c], ar);
    }
    xl[t] = al; xr[t] = ar;
}

// ---------------- conv2 edge passes (1 head, 32 channels) ----------------
__global__ void k_score2(const int* __restrict__ ei,
                         const float* __restrict__ xl, const float* __restrict__ xr,
                         const float* __restrict__ att,
                         float* __restrict__ e2, unsigned* __restrict__ m2) {
    int e = blockIdx.x * blockDim.x + threadIdx.x;     // ETOT threads
    int s, d; edge_pair(ei, e, s, d);
    const float* pl = xl + s * OUT_F;
    const float* pr = xr + d * OUT_F;
    float acc = 0.0f;
#pragma unroll
    for (int c = 0; c < OUT_F; ++c)
        acc = fmaf(att[c], lrelu(pl[c] + pr[c]), acc);
    e2[e] = acc;
    atomicMax(&m2[d], fkey(acc));
}

__global__ void k_exp2(const int* __restrict__ ei,
                       float* __restrict__ e2, const unsigned* __restrict__ m2,
                       float* __restrict__ s2) {
    int e = blockIdx.x * blockDim.x + threadIdx.x;
    int s, d; edge_pair(ei, e, s, d);
    (void)s;
    float p = __expf(e2[e] - fdecode(m2[d]));
    e2[e] = p;
    atomicAdd(&s2[d], p);
}

__global__ void k_agg2(const int* __restrict__ ei,
                       const float* __restrict__ e2, const float* __restrict__ s2,
                       const float* __restrict__ xl, float* __restrict__ h2) {
    int e = blockIdx.x * blockDim.x + threadIdx.x;
    int s, d; edge_pair(ei, e, s, d);
    float alpha = e2[e] / (s2[d] + 1e-16f);
    const float* pl = xl + s * OUT_F;
    float* po = h2 + d * OUT_F;
#pragma unroll
    for (int c = 0; c < OUT_F; ++c)
        atomicAdd(&po[c], alpha * pl[c]);
}

// ---------------- bias + L2 row normalize; emit z (f32 out, f16 for WMMA) ---
__global__ void k_final(const float* __restrict__ h2, const float* __restrict__ bias2,
                        float* __restrict__ zout, _Float16* __restrict__ zh) {
    int n = blockIdx.x * blockDim.x + threadIdx.x;     // N threads
    float v[OUT_F];
    float ss = 0.0f;
#pragma unroll
    for (int c = 0; c < OUT_F; ++c) {
        v[c] = h2[n * OUT_F + c] + bias2[c];
        ss = fmaf(v[c], v[c], ss);
    }
    float inv = 1.0f / fmaxf(sqrtf(ss), 1e-12f);
#pragma unroll
    for (int c = 0; c < OUT_F; ++c) {
        float z = v[c] * inv;
        zout[n * OUT_F + c] = z;
        zh[n * OUT_F + c] = (_Float16)z;
    }
}

// ---------------- A_pred = sigmoid(z @ z^T) via V_WMMA_F32_16X16X32_F16 -----
// One wave holds a 16x32 A fragment (its i-tile) and sweeps 8 j-tiles,
// unrolled x2 so two independent B-load/WMMA/epilogue chains overlap.
// Block: 256 threads (8 waves). Grid: (512 i-tiles, 8).
__global__ void __launch_bounds__(256) k_adj(const _Float16* __restrict__ zh,
                                             float* __restrict__ out) {
    const int lane = threadIdx.x & 31;
    const int wave = threadIdx.x >> 5;
    const int lmod = lane & 15;
    const int hi   = (lane >> 4);                 // 0 for lanes 0-15, 1 for 16-31

    // ---- A fragment: 16x32 f16, rows i0..i0+15 ----
    const int i0    = blockIdx.x * 16;
    const int rowA  = i0 + lmod;
    const int kbA   = hi ? 16 : 0;                // lanes 16-31 carry K=16..31
    v16h a = *(const v16h*)(zh + rowA * 32 + kbA);       // K = kbA .. kbA+15

    const int kbB   = hi ? 16 : 0;                // B 32x16: lanes 0-15 K=0..15
    const int jBase = (blockIdx.y * 8 + wave) * 8;
    const int mbase = i0 + (hi ? 8 : 0);

#pragma unroll 2
    for (int jj = 0; jj < 8; ++jj) {
        const int j0   = (jBase + jj) * 16;
        const int rowB = j0 + lmod;
        // keep the next B fragment warm at WGP scope (z is L2-resident)
        __builtin_prefetch(zh + (rowB + 16) * 32 + kbB, 0, 3);
        v16h b = *(const v16h*)(zh + rowB * 32 + kbB);   // K = kbB .. kbB+15

        v8f c = {};
        c = __builtin_amdgcn_wmma_f32_16x16x32_f16(
                /*neg_a=*/false, a, /*neg_b=*/false, b,
                /*c_mod=*/(short)0, c, /*reuse_a=*/false, /*reuse_b=*/false);

        const size_t col = (size_t)(j0 + lmod);
#pragma unroll
        for (int r = 0; r < 8; ++r) {
            float sg = fast_sigmoid(c[r]);
            // write-once 256MB stream: NT hint keeps it out of L2
            __builtin_nontemporal_store(sg, &out[(size_t)(mbase + r) * N_NODES + col]);
        }
    }
}

// ---------------------------------------------------------------------------
extern "C" void kernel_launch(void* const* d_in, const int* in_sizes, int n_in,
                              void* d_out, int out_size, void* d_ws, size_t ws_size,
                              hipStream_t stream) {
    const float* x     = (const float*)d_in[0];
    const int*   ei    = (const int*)  d_in[1];
    const float* W1l   = (const float*)d_in[2];
    const float* b1l   = (const float*)d_in[3];
    const float* W1r   = (const float*)d_in[4];
    const float* b1r   = (const float*)d_in[5];
    const float* att1  = (const float*)d_in[6];
    const float* bias1 = (const float*)d_in[7];
    const float* W2l   = (const float*)d_in[8];
    const float* b2l   = (const float*)d_in[9];
    const float* W2r   = (const float*)d_in[10];
    const float* b2r   = (const float*)d_in[11];
    const float* att2  = (const float*)d_in[12];
    const float* bias2 = (const float*)d_in[13];

    char* ws = (char*)d_ws;
    size_t off = 0;
    // --- zero-init accumulator region (one memset) ---
    float*    h1 = (float*)   (ws + off); off += (size_t)N_NODES * F1 * 4;      // 4 MB
    unsigned* m1 = (unsigned*)(ws + off); off += (size_t)N_NODES * HEADS * 4;   // 256 KB
    float*    s1 = (float*)   (ws + off); off += (size_t)N_NODES * HEADS * 4;   // 256 KB
    float*    h2 = (float*)   (ws + off); off += (size_t)N_NODES * OUT_F * 4;   // 1 MB
    unsigned* m2 = (unsigned*)(ws + off); off += (size_t)N_NODES * 4;           // 32 KB
    float*    s2 = (float*)   (ws + off); off += (size_t)N_NODES * 4;           // 32 KB
    const size_t zero_bytes = off;
    // --- non-zeroed scratch ---
    float* xl1 = (float*)(ws + off); off += (size_t)N_NODES * F1 * 4;
    float* xr1 = (float*)(ws + off); off += (size_t)N_NODES * F1 * 4;
    float* e1  = (float*)(ws + off); off += (size_t)ETOT * HEADS * 4;
    float* xl2 = (float*)(ws + off); off += (size_t)N_NODES * OUT_F * 4;
    float* xr2 = (float*)(ws + off); off += (size_t)N_NODES * OUT_F * 4;
    float* e2  = (float*)(ws + off); off += (size_t)ETOT * 4;
    off = (off + 63) & ~(size_t)63;                  // 64B align for f16 vector loads
    _Float16* zh = (_Float16*)(ws + off); off += (size_t)N_NODES * OUT_F * 2;

    float* A_pred = (float*)d_out;                                   // [N, N]
    float* z_out  = (float*)d_out + (size_t)N_NODES * N_NODES;       // [N, 32]

    hipMemsetAsync(d_ws, 0, zero_bytes, stream);

    // conv1
    k_linear1<<<(N_NODES * F1) / 256, 256, 0, stream>>>(x, W1l, b1l, W1r, b1r, xl1, xr1);
    k_score1 <<<(ETOT * HEADS) / 256, 256, 0, stream>>>(ei, xl1, xr1, att1, e1, m1);
    k_exp1   <<<(ETOT * HEADS) / 256, 256, 0, stream>>>(ei, e1, m1, s1);
    k_agg1   <<<(ETOT * HEADS) / 256, 256, 0, stream>>>(ei, e1, s1, xl1, h1);
    // conv2
    k_linear2<<<(N_NODES * OUT_F) / 256, 256, 0, stream>>>(h1, bias1, W2l, b2l, W2r, b2r, xl2, xr2);
    k_score2 <<<ETOT / 256, 256, 0, stream>>>(ei, xl2, xr2, att2, e2, m2);
    k_exp2   <<<ETOT / 256, 256, 0, stream>>>(ei, e2, m2, s2);
    k_agg2   <<<ETOT / 256, 256, 0, stream>>>(ei, e2, s2, xl2, h2);
    // normalize -> z (f32 to output, f16 for WMMA)
    k_final  <<<N_NODES / 256, 256, 0, stream>>>(h2, bias2, z_out, zh);
    // A_pred = sigmoid(z z^T): 512 i-tiles x (8 blocks * 8 waves * 8 tiles) j
    k_adj    <<<dim3(512, 8), 256, 0, stream>>>(zh, A_pred);
}
